// RNN_13666585936059
// MI455X (gfx1250) — matrix operand order repaired and checked by reading, not compiled, not run on hardware
//
#include <hip/hip_runtime.h>
#include <hip/hip_bf16.h>
#include <math.h>

// Problem dims (match reference)
#define B_DIM 64
#define T_DIM 1024
#define E_DIM 512
#define H_DIM 1024
#define V_DIM 512
#define RNN_NWG 32   // persistent-recurrence workgroups (32 cols of H each)

typedef __bf16 bf16_t;
typedef __attribute__((ext_vector_type(16))) __bf16 v16bf;
typedef __attribute__((ext_vector_type(8)))  __bf16 v8bf;
typedef __attribute__((ext_vector_type(8)))  float  v8f;

// ---- helpers -------------------------------------------------------------

// Native hardware f32->bf16 conversion (RNE).
static __device__ __forceinline__ bf16_t f2bf(float f) { return (bf16_t)f; }

// A fragment (16x32 bf16), contiguous k order (not the ISA table order):
// elems 0..15 -> K = k0+16*half .. +15 (contiguous, matches B)
static __device__ __forceinline__ v16bf load_fragA(const bf16_t* __restrict__ row,
                                                   int k0, int half) {
  const bf16_t* p = row + k0 + half * 16;
  v8bf lo = *(const v8bf*)p;          // 16B
  v8bf hi = *(const v8bf*)(p + 8);    // 16B
  v16bf r;
#pragma unroll
  for (int i = 0; i < 8; ++i) { r[i] = lo[i]; r[i + 8] = hi[i]; }
  return r;
}

// B fragment (32x16 bf16): lane = N, K linear: elem i -> K = k0+16*half+i.
// Weight rows are N-major, so this is one contiguous 32B load (works for
// both global and LDS source pointers).
template <typename P>
static __device__ __forceinline__ v16bf load_fragB(const P* row, int k0, int half) {
  return *(const v16bf*)(row + k0 + half * 16);
}

static __device__ __forceinline__ v8f wmma_bf16(v16bf a, v16bf b, v8f c) {
  return __builtin_amdgcn_wmma_f32_16x16x32_bf16(false, a, false, b, (short)0, c, false, false);
}

// ---- prep kernels --------------------------------------------------------

// elementwise f32 -> bf16 (for streaming X)
__global__ void convert_bf16(const float* __restrict__ in, bf16_t* __restrict__ out, int total) {
  int idx = blockIdx.x * blockDim.x + threadIdx.x;
  if (idx < total) out[idx] = f2bf(in[idx]);
}

// in: K x N f32 (row-major)  ->  out: N x K bf16 (row-major)
__global__ void transpose_to_bf16(const float* __restrict__ in, bf16_t* __restrict__ out,
                                  int K, int N) {
  int idx = blockIdx.x * blockDim.x + threadIdx.x;
  int total = K * N;
  if (idx < total) {
    int k = idx / N, n = idx % N;
    out[(size_t)n * K + k] = f2bf(in[idx]);
  }
}

__global__ void init_state(bf16_t* __restrict__ h0, unsigned* __restrict__ counter, int count) {
  int idx = blockIdx.x * blockDim.x + threadIdx.x;
  if (idx < count) h0[idx] = f2bf(0.0f);
  if (idx == 0) *counter = 0u;
}

// ---- generic GEMM: C = A(bf16, MxK) @ Bt(bf16, NxK)^T + bias, C f32 MxN ---
// 256 threads = 8 waves, arranged 2(M) x 4(N); each wave computes a 32x32
// output block as 2x2 WMMA tiles (4 accumulators) -> 32 B of operand loads
// per v_wmma.
__global__ __launch_bounds__(256) void gemm_bf16(const bf16_t* __restrict__ A,
                                                 const bf16_t* __restrict__ Bt,
                                                 const float* __restrict__ bias,
                                                 float* __restrict__ C,
                                                 int K, int N) {
  const int lane = threadIdx.x & 31;
  const int wave = threadIdx.x >> 5;
  const int waveM = wave & 1, waveN = wave >> 1;
  const int half = lane >> 4, l15 = lane & 15;
  const int m0 = blockIdx.x * 64 + waveM * 32;    // wave's first M tile
  const int n0 = blockIdx.y * 128 + waveN * 32;   // wave's first N tile

  const bf16_t* Ar0 = A  + (size_t)(m0 + l15) * K;
  const bf16_t* Ar1 = A  + (size_t)(m0 + 16 + l15) * K;
  const bf16_t* Br0 = Bt + (size_t)(n0 + l15) * K;
  const bf16_t* Br1 = Bt + (size_t)(n0 + 16 + l15) * K;

  v8f c00 = {}, c01 = {}, c10 = {}, c11 = {};
  for (int k0 = 0; k0 < K; k0 += 32) {
    if (k0 + 32 < K) {
      __builtin_prefetch(Ar0 + k0 + 32, 0, 3);
      __builtin_prefetch(Ar1 + k0 + 32, 0, 3);
    }
    v16bf a0 = load_fragA(Ar0, k0, half);
    v16bf a1 = load_fragA(Ar1, k0, half);
    v16bf b0 = load_fragB(Br0, k0, half);
    v16bf b1 = load_fragB(Br1, k0, half);
    c00 = wmma_bf16(a0, b0, c00);
    c01 = wmma_bf16(a0, b1, c01);
    c10 = wmma_bf16(a1, b0, c10);
    c11 = wmma_bf16(a1, b1, c11);
  }

  const int nA = n0 + l15, nB = nA + 16;
  const float biasA = bias[nA], biasB = bias[nB];
#pragma unroll
  for (int r = 0; r < 8; ++r) {
    const int mA = m0 + r + 8 * half;
    const int mB = mA + 16;
    C[(size_t)mA * N + nA] = c00[r] + biasA;
    C[(size_t)mA * N + nB] = c01[r] + biasB;
    C[(size_t)mB * N + nA] = c10[r] + biasA;
    C[(size_t)mB * N + nB] = c11[r] + biasB;
  }
}

// ---- persistent recurrence -----------------------------------------------
// 32 WGs of 128 threads (4 waves). WG g owns H columns [32g, 32g+32); its
// W_hh^T slice (32 x 1024 bf16 = 64 KB) is staged into LDS once and read via
// ds for all 1024 steps. Wave w owns batch rows [16w, 16w+16) and computes
// BOTH 16-col N tiles (2 WMMAs sharing one A fragment) -> h_prev is read
// exactly once per WG per step (no duplication). Device-wide barrier per
// step via an agent-scope atomic counter.
__global__ __launch_bounds__(128) void rnn_steps(const float* __restrict__ xproj,
                                                 const bf16_t* __restrict__ WhhT,
                                                 bf16_t* __restrict__ hbuf0,
                                                 bf16_t* __restrict__ hbuf1,
                                                 bf16_t* __restrict__ hs,
                                                 float* __restrict__ hfinal,
                                                 unsigned* __restrict__ counter) {
  __shared__ __align__(16) bf16_t ldsW[32 * H_DIM];   // 64 KB

  // cooperative 16B-vector fill of the weight slice
  {
    const uint4* src = (const uint4*)(WhhT + (size_t)blockIdx.x * 32 * H_DIM);
    uint4* dst = (uint4*)ldsW;
    for (int i = threadIdx.x; i < (32 * H_DIM * 2) / 16; i += 128) dst[i] = src[i];
  }
  __syncthreads();

  const int lane = threadIdx.x & 31;
  const int mt = threadIdx.x >> 5;                  // wave id = M tile 0..3
  const int half = lane >> 4, l15 = lane & 15;
  const int n0 = blockIdx.x * 32 + l15;             // global H column, tile 0
  const int n1 = n0 + 16;                           // global H column, tile 1
  const bf16_t* BrowL0 = ldsW + (size_t)l15 * H_DIM;
  const bf16_t* BrowL1 = ldsW + (size_t)(16 + l15) * H_DIM;

  for (int t = 0; t < T_DIM; ++t) {
    const bf16_t* hPrev = (t & 1) ? hbuf1 : hbuf0;
    bf16_t*       hNext = (t & 1) ? hbuf0 : hbuf1;
    const bf16_t* Arow  = hPrev + (size_t)(mt * 16 + l15) * H_DIM;

    v8f c0 = {}, c1 = {};
    for (int k0 = 0; k0 < H_DIM; k0 += 32) {
      v16bf a  = load_fragA(Arow, k0, half);
      v16bf b0 = load_fragB(BrowL0, k0, half);      // from LDS
      v16bf b1 = load_fragB(BrowL1, k0, half);      // from LDS
      c0 = wmma_bf16(a, b0, c0);
      c1 = wmma_bf16(a, b1, c1);
    }

#pragma unroll
    for (int r = 0; r < 8; ++r) {
      const int m = mt * 16 + r + 8 * half;         // batch row 0..63
      const size_t bt = (size_t)m * T_DIM + t;
      float v0 = tanhf(xproj[bt * H_DIM + n0] + c0[r]);
      float v1 = tanhf(xproj[bt * H_DIM + n1] + c1[r]);
      bf16_t bv0 = f2bf(v0), bv1 = f2bf(v1);
      hNext[(size_t)m * H_DIM + n0] = bv0;
      hNext[(size_t)m * H_DIM + n1] = bv1;
      hs[bt * H_DIM + n0] = bv0;
      hs[bt * H_DIM + n1] = bv1;
      if (t + 1 < T_DIM) __builtin_prefetch(&xproj[(bt + 1) * H_DIM + n0], 0, 0);
      if (t == T_DIM - 1) {
        hfinal[(size_t)m * H_DIM + n0] = v0;
        hfinal[(size_t)m * H_DIM + n1] = v1;
      }
    }

    // ---- device-wide barrier between timesteps ----
    __threadfence();
    __syncthreads();
    if (threadIdx.x == 0) {
      __hip_atomic_fetch_add(counter, 1u, __ATOMIC_RELEASE, __HIP_MEMORY_SCOPE_AGENT);
      const unsigned target = (unsigned)RNN_NWG * (unsigned)(t + 1);
      while (__hip_atomic_load(counter, __ATOMIC_ACQUIRE, __HIP_MEMORY_SCOPE_AGENT) < target) {
        __builtin_amdgcn_s_sleep(2);
      }
    }
    __syncthreads();
    __threadfence();
  }
}

// ---- host glue -----------------------------------------------------------

extern "C" void kernel_launch(void* const* d_in, const int* in_sizes, int n_in,
                              void* d_out, int out_size, void* d_ws, size_t ws_size,
                              hipStream_t stream) {
  const float* X    = (const float*)d_in[0];   // (B,T,E)
  const float* W_xh = (const float*)d_in[1];   // (E,H)
  const float* b_h  = (const float*)d_in[2];   // (H)
  const float* W_hh = (const float*)d_in[3];   // (H,H)
  const float* W_hy = (const float*)d_in[4];   // (H,V)
  const float* b_y  = (const float*)d_in[5];   // (V)

  float* Y      = (float*)d_out;                               // (B,T,V)
  float* hfinal = Y + (size_t)B_DIM * T_DIM * V_DIM;           // (B,H)

  // Workspace carve-out (all chunks are multiples of 256B)
  char* ws = (char*)d_ws;
  float*  xproj = (float*)ws;   ws += (size_t)B_DIM * T_DIM * H_DIM * sizeof(float);
  bf16_t* hs    = (bf16_t*)ws;  ws += (size_t)B_DIM * T_DIM * H_DIM * sizeof(bf16_t);
  bf16_t* Xbf   = (bf16_t*)ws;  ws += (size_t)B_DIM * T_DIM * E_DIM * sizeof(bf16_t);
  bf16_t* WxhT  = (bf16_t*)ws;  ws += (size_t)H_DIM * E_DIM * sizeof(bf16_t);
  bf16_t* WhhT  = (bf16_t*)ws;  ws += (size_t)H_DIM * H_DIM * sizeof(bf16_t);
  bf16_t* WhyT  = (bf16_t*)ws;  ws += (size_t)V_DIM * H_DIM * sizeof(bf16_t);
  bf16_t* h0    = (bf16_t*)ws;  ws += (size_t)B_DIM * H_DIM * sizeof(bf16_t);
  bf16_t* h1    = (bf16_t*)ws;  ws += (size_t)B_DIM * H_DIM * sizeof(bf16_t);
  unsigned* counter = (unsigned*)ws;

  // Prep: convert X, transpose weights to N-major bf16, zero h0 + counter.
  {
    int n = B_DIM * T_DIM * E_DIM;
    convert_bf16<<<(n + 255) / 256, 256, 0, stream>>>(X, Xbf, n);
  }
  {
    int n = E_DIM * H_DIM;
    transpose_to_bf16<<<(n + 255) / 256, 256, 0, stream>>>(W_xh, WxhT, E_DIM, H_DIM);
  }
  {
    int n = H_DIM * H_DIM;
    transpose_to_bf16<<<(n + 255) / 256, 256, 0, stream>>>(W_hh, WhhT, H_DIM, H_DIM);
  }
  {
    int n = H_DIM * V_DIM;
    transpose_to_bf16<<<(n + 255) / 256, 256, 0, stream>>>(W_hy, WhyT, H_DIM, V_DIM);
  }
  {
    int n = B_DIM * H_DIM;
    init_state<<<(n + 255) / 256, 256, 0, stream>>>(h0, counter, n);
  }

  // Phase 1: xproj = X @ W_xh + b_h  (M=65536, K=512, N=1024)
  gemm_bf16<<<dim3((B_DIM * T_DIM) / 64, H_DIM / 128), 256, 0, stream>>>(
      Xbf, WxhT, b_h, xproj, E_DIM, H_DIM);

  // Phase 2: persistent recurrence (32 co-resident WGs, barrier per step)
  rnn_steps<<<RNN_NWG, 128, 0, stream>>>(xproj, WhhT, h0, h1, hs, hfinal, counter);

  // Phase 3: Y = hs @ W_hy + b_y  (M=65536, K=1024, N=512)
  gemm_bf16<<<dim3((B_DIM * T_DIM) / 64, V_DIM / 128), 256, 0, stream>>>(
      hs, WhyT, b_y, Y, H_DIM, V_DIM);
}
